// GatedAttention_68822555951691
// MI455X (gfx1250) — matrix operand order repaired and checked by reading, not compiled, not run on hardware
//
#include <hip/hip_runtime.h>

#define DIM 512
#define HEADS 8
#define HD 64
#define BATCH 2
#define SEQ 4096
#define MROWS (BATCH * SEQ) /* 8192 */
#define NC3 (3 * DIM)       /* 1536 */
#define SM_SCALE 0.125f     /* 64^-0.5 */

typedef __attribute__((ext_vector_type(16))) __bf16 v16bf;
typedef __attribute__((ext_vector_type(8)))  float  v8f;

union Frag {
  v16bf v;
  unsigned int u[8];
};

__device__ __forceinline__ unsigned short f2bf(float f) {
  union { float f; unsigned int u; } c; c.f = f;
  unsigned int r = c.u + 0x7FFFu + ((c.u >> 16) & 1u);
  return (unsigned short)(r >> 16);
}

// A-fragment K index for VGPR j, lane-half hi (16-bit A 16x32 layout)
__device__ __forceinline__ int kkA(int j, int hi) {
  return (j < 4) ? (hi * 8 + 2 * j) : (16 + hi * 8 + 2 * (j - 4));
}
// B-fragment K index (same K order as A: VGPR 0-3 K=hi*8.., VGPR 4-7 K=16+hi*8..)
__device__ __forceinline__ int kkB(int j, int hi) {
  return (j < 4) ? (hi * 8 + 2 * j) : (16 + hi * 8 + 2 * (j - 4));
}

// CDNA5 async global->LDS copy (ASYNCcnt tracked, no VGPR round-trip).
// vdst = LDS byte offset (low 32 bits of the generic pointer = LDS offset).
__device__ __forceinline__ void async_load_b128(unsigned lds_off, const void* gaddr) {
  asm volatile("global_load_async_to_lds_b128 %0, %1, off"
               :: "v"(lds_off), "v"(gaddr)
               : "memory");
}
__device__ __forceinline__ void wait_async0() {
  asm volatile("s_wait_asynccnt 0" ::: "memory");
}

// ---------------------------------------------------------------- convert
__global__ void convert_kernel(const float* __restrict__ x,
                               const float* __restrict__ wq,
                               const float* __restrict__ wp,
                               unsigned short* __restrict__ xb,
                               unsigned short* __restrict__ wqb,
                               unsigned short* __restrict__ wpb) {
  int stride = gridDim.x * blockDim.x;
  int t = blockIdx.x * blockDim.x + threadIdx.x;
  for (int i = t; i < MROWS * DIM; i += stride) xb[i] = f2bf(x[i]);
  for (int i = t; i < DIM * NC3; i += stride) wqb[i] = f2bf(wq[i]);
  for (int i = t; i < DIM * DIM; i += stride) wpb[i] = f2bf(wp[i]);
}

// ---------------------------------------------------------------- gate
__global__ void gate_kernel(const float* __restrict__ x,
                            const float* __restrict__ gw,
                            const float* __restrict__ gb,
                            float* __restrict__ gate) {
  int idx = blockIdx.x * blockDim.x + threadIdx.x; // [0, 65536)
  int row = idx >> 3;
  int h = idx & 7;
  float acc = 0.f;
  for (int c = 0; c < DIM; ++c)
    acc += x[(size_t)row * DIM + c] * gw[c * HEADS + h];
  acc += gb[h];
  gate[idx] = 1.f / (1.f + __expf(-acc));
}

// ---------------------------------------------------------------- QKV GEMM
// C[m,n] = sum_k xb[m,k]*wq[k,n] + bias[n], scattered into q/k/v [B,H,N,hd].
// Double-buffered LDS; A tiles via async-to-LDS; B tiles pipelined via VGPRs.
__global__ __launch_bounds__(128) void qkv_gemm_kernel(
    const unsigned short* __restrict__ xb,
    const unsigned short* __restrict__ wq,
    const float* __restrict__ bias,
    unsigned short* __restrict__ qb,
    unsigned short* __restrict__ kb,
    unsigned short* __restrict__ vb) {
  __shared__ unsigned short As[2][64 * 40];
  __shared__ unsigned short Bs[2][64 * 40];
  const int tid = threadIdx.x;
  const int wave = tid >> 5, lane = tid & 31;
  const int lo = lane & 15, hi = lane >> 4;
  const int nbase = blockIdx.x * 64;
  const int mbase = blockIdx.y * 64;

  // A staging: 256 16B chunks, 2 per thread. chunk c: row=c>>2, col8=(c&3)*8
  const int ar0 = tid >> 2, ac = (tid & 3) * 8;
  const int ar1 = ar0 + 32;
  // B staging: thread covers W row kr, 16 cols from nb2
  const int kr = tid >> 2, nb2 = (tid & 3) * 16;

  uint4 breg0, breg1;

  auto issueA = [&](int k0, int buf) {
    async_load_b128((unsigned)(size_t)&As[buf][ar0 * 40 + ac],
                    xb + (size_t)(mbase + ar0) * DIM + k0 + ac);
    async_load_b128((unsigned)(size_t)&As[buf][ar1 * 40 + ac],
                    xb + (size_t)(mbase + ar1) * DIM + k0 + ac);
  };
  auto loadB = [&](int k0) {
    const unsigned short* wsrc = wq + (size_t)(k0 + kr) * NC3 + nbase + nb2;
    breg0 = *(const uint4*)wsrc;
    breg1 = *(const uint4*)(wsrc + 8);
  };
  auto storeB = [&](int buf) {
    alignas(16) unsigned short tmp[16];
    *(uint4*)&tmp[0] = breg0;
    *(uint4*)&tmp[8] = breg1;
#pragma unroll
    for (int i = 0; i < 16; ++i) Bs[buf][(nb2 + i) * 40 + kr] = tmp[i];
  };

  const v8f vzero = {0, 0, 0, 0, 0, 0, 0, 0};
  v8f acc[4];
  acc[0] = acc[1] = acc[2] = acc[3] = vzero;

  issueA(0, 0);
  loadB(0);
  storeB(0);
  wait_async0();
  __syncthreads();

  for (int it = 0; it < 16; ++it) {
    const int cur = it & 1;
    if (it < 15) {
      issueA((it + 1) * 32, cur ^ 1);
      loadB((it + 1) * 32);
    }

    Frag a;
#pragma unroll
    for (int j = 0; j < 8; ++j)
      a.u[j] = *(const unsigned int*)&As[cur][(wave * 16 + lo) * 40 + kkA(j, hi)];
    Frag bf[4];
#pragma unroll
    for (int nb = 0; nb < 4; ++nb)
#pragma unroll
      for (int j = 0; j < 8; ++j)
        bf[nb].u[j] = *(const unsigned int*)&Bs[cur][(nb * 16 + lo) * 40 + kkB(j, hi)];
#pragma unroll
    for (int nb = 0; nb < 4; ++nb)
      acc[nb] = __builtin_amdgcn_wmma_f32_16x16x32_bf16(
          false, a.v, false, bf[nb].v, (short)0, acc[nb], false, false);

    if (it < 15) {
      storeB(cur ^ 1);
      wait_async0();
    }
    __syncthreads();
  }

  const int which = nbase >> 9;   // 0=q,1=k,2=v (64-col block stays inside one)
  const int h = (nbase >> 6) & 7; // head, constant per block
  unsigned short* dst = (which == 0) ? qb : (which == 1) ? kb : vb;
#pragma unroll
  for (int nb = 0; nb < 4; ++nb) {
    int col = nbase + nb * 16 + lo;
    float bv = bias[col];
    int d = nb * 16 + lo;
#pragma unroll
    for (int r = 0; r < 8; ++r) {
      int m = mbase + wave * 16 + hi * 8 + r;
      int bidx = m >> 12;
      int nidx = m & (SEQ - 1);
      dst[((size_t)(bidx * HEADS + h) * SEQ + nidx) * HD + d] = f2bf(acc[nb][r] + bv);
    }
  }
}

// ---------------------------------------------------------------- attention
// Flash-attention per (b, h, 64-query tile). 4 waves, 16 query rows per wave.
// Q/K staged via async-to-LDS; V transposed through VGPRs.
__global__ __launch_bounds__(128) void attn_kernel(
    const unsigned short* __restrict__ qbuf,
    const unsigned short* __restrict__ kbuf,
    const unsigned short* __restrict__ vbuf,
    const float* __restrict__ gate,
    unsigned short* __restrict__ yb) {
  __shared__ unsigned short Qs[64 * 72];
  __shared__ unsigned short Ks[64 * 72];
  __shared__ unsigned short Vs[64 * 72]; // transposed: [dim][key]
  __shared__ unsigned short Ps[64 * 72]; // per-wave 16-row slices

  const int tid = threadIdx.x;
  const int wave = tid >> 5, lane = tid & 31;
  const int lo = lane & 15, hi = lane >> 4;
  const int h = blockIdx.y;
  const int b = blockIdx.z;
  const int bh = b * HEADS + h;
  const int rowbase = blockIdx.x * 64;
  const size_t headoff = (size_t)bh * SEQ * HD;

  // async-stage Q tile (64x64 bf16 = 512 x 16B chunks, 4 per thread)
#pragma unroll
  for (int i = 0; i < 4; ++i) {
    int c = tid + i * 128;
    int row = c >> 3, cb = (c & 7) * 8;
    async_load_b128((unsigned)(size_t)&Qs[row * 72 + cb],
                    qbuf + headoff + (size_t)(rowbase + row) * HD + cb);
  }

  const v8f vzero = {0, 0, 0, 0, 0, 0, 0, 0};
  v8f o[4];
  o[0] = o[1] = o[2] = o[3] = vzero;
  float mrow[8], lrow[8];
#pragma unroll
  for (int r = 0; r < 8; ++r) { mrow[r] = -1e30f; lrow[r] = 0.f; }

  for (int kb0 = 0; kb0 < SEQ; kb0 += 64) {
    __syncthreads(); // all waves done reading Ks/Vs from previous chunk
    // async-stage K chunk
#pragma unroll
    for (int i = 0; i < 4; ++i) {
      int c = tid + i * 128;
      int row = c >> 3, cb = (c & 7) * 8;
      async_load_b128((unsigned)(size_t)&Ks[row * 72 + cb],
                      kbuf + headoff + (size_t)(kb0 + row) * HD + cb);
    }
    { // stage V transposed via VGPRs
      int row = tid >> 1;
      int db = (tid & 1) * 32;
      const unsigned short* vsrc = vbuf + headoff + (size_t)(kb0 + row) * HD + db;
      alignas(16) unsigned short tmp[32];
      *(uint4*)&tmp[0]  = *(const uint4*)(vsrc);
      *(uint4*)&tmp[8]  = *(const uint4*)(vsrc + 8);
      *(uint4*)&tmp[16] = *(const uint4*)(vsrc + 16);
      *(uint4*)&tmp[24] = *(const uint4*)(vsrc + 24);
#pragma unroll
      for (int i = 0; i < 32; ++i) Vs[(db + i) * 72 + row] = tmp[i];
    }
    wait_async0(); // covers this chunk's K (and first-iteration Q)
    __syncthreads();

    // S = Q @ K^T  (16 rows x 64 keys per wave)
    v8f s[4];
    s[0] = s[1] = s[2] = s[3] = vzero;
#pragma unroll
    for (int st = 0; st < 2; ++st) {
      Frag aq;
#pragma unroll
      for (int j = 0; j < 8; ++j)
        aq.u[j] = *(const unsigned int*)&Qs[(wave * 16 + lo) * 72 + st * 32 + kkA(j, hi)];
      Frag bk[4];
#pragma unroll
      for (int nb = 0; nb < 4; ++nb)
#pragma unroll
        for (int j = 0; j < 8; ++j)
          bk[nb].u[j] = *(const unsigned int*)&Ks[(nb * 16 + lo) * 72 + st * 32 + kkB(j, hi)];
#pragma unroll
      for (int nb = 0; nb < 4; ++nb)
        s[nb] = __builtin_amdgcn_wmma_f32_16x16x32_bf16(
            false, aq.v, false, bk[nb].v, (short)0, s[nb], false, false);
    }

    // online softmax; a row lives across the 16 lanes sharing `hi`
#pragma unroll
    for (int r = 0; r < 8; ++r) {
      float mx = fmaxf(fmaxf(s[0][r], s[1][r]), fmaxf(s[2][r], s[3][r]));
      mx = fmaxf(mx, __shfl_xor(mx, 1, 32));
      mx = fmaxf(mx, __shfl_xor(mx, 2, 32));
      mx = fmaxf(mx, __shfl_xor(mx, 4, 32));
      mx = fmaxf(mx, __shfl_xor(mx, 8, 32));
      float mnew = fmaxf(mrow[r], mx * SM_SCALE);
      float alpha = __expf(mrow[r] - mnew);
      float rsum = 0.f;
#pragma unroll
      for (int nb = 0; nb < 4; ++nb) {
        float p = __expf(s[nb][r] * SM_SCALE - mnew);
        rsum += p;
        Ps[(wave * 16 + hi * 8 + r) * 72 + nb * 16 + lo] = f2bf(p);
        o[nb][r] *= alpha;
      }
      rsum += __shfl_xor(rsum, 1, 32);
      rsum += __shfl_xor(rsum, 2, 32);
      rsum += __shfl_xor(rsum, 4, 32);
      rsum += __shfl_xor(rsum, 8, 32);
      lrow[r] = lrow[r] * alpha + rsum;
      mrow[r] = mnew;
    }

    // O += P @ V
#pragma unroll
    for (int st = 0; st < 2; ++st) {
      Frag ap;
#pragma unroll
      for (int j = 0; j < 8; ++j)
        ap.u[j] = *(const unsigned int*)&Ps[(wave * 16 + lo) * 72 + st * 32 + kkA(j, hi)];
      Frag bv[4];
#pragma unroll
      for (int nb = 0; nb < 4; ++nb)
#pragma unroll
        for (int j = 0; j < 8; ++j)
          bv[nb].u[j] = *(const unsigned int*)&Vs[(nb * 16 + lo) * 72 + st * 32 + kkB(j, hi)];
#pragma unroll
      for (int nb = 0; nb < 4; ++nb)
        o[nb] = __builtin_amdgcn_wmma_f32_16x16x32_bf16(
            false, ap.v, false, bv[nb].v, (short)0, o[nb], false, false);
    }
  }

  // normalize, gate, store y[b, n, h*64+d] as bf16
#pragma unroll
  for (int r = 0; r < 8; ++r) {
    int row = rowbase + wave * 16 + hi * 8 + r;
    float inv = 1.f / lrow[r];
    float g = gate[((size_t)b * SEQ + row) * HEADS + h];
    float gi = inv * g;
#pragma unroll
    for (int nb = 0; nb < 4; ++nb)
      yb[((size_t)b * SEQ + row) * DIM + h * HD + nb * 16 + lo] = f2bf(o[nb][r] * gi);
  }
}

// ---------------------------------------------------------------- proj GEMM
__global__ __launch_bounds__(128) void proj_gemm_kernel(
    const unsigned short* __restrict__ yb,
    const unsigned short* __restrict__ wp,
    const float* __restrict__ bias,
    float* __restrict__ out) {
  __shared__ unsigned short As[2][64 * 40];
  __shared__ unsigned short Bs[2][64 * 40];
  const int tid = threadIdx.x;
  const int wave = tid >> 5, lane = tid & 31;
  const int lo = lane & 15, hi = lane >> 4;
  const int nbase = blockIdx.x * 64;
  const int mbase = blockIdx.y * 64;

  const int ar0 = tid >> 2, ac = (tid & 3) * 8;
  const int ar1 = ar0 + 32;
  const int kr = tid >> 2, nb2 = (tid & 3) * 16;

  uint4 breg0, breg1;

  auto issueA = [&](int k0, int buf) {
    async_load_b128((unsigned)(size_t)&As[buf][ar0 * 40 + ac],
                    yb + (size_t)(mbase + ar0) * DIM + k0 + ac);
    async_load_b128((unsigned)(size_t)&As[buf][ar1 * 40 + ac],
                    yb + (size_t)(mbase + ar1) * DIM + k0 + ac);
  };
  auto loadB = [&](int k0) {
    const unsigned short* wsrc = wp + (size_t)(k0 + kr) * DIM + nbase + nb2;
    breg0 = *(const uint4*)wsrc;
    breg1 = *(const uint4*)(wsrc + 8);
  };
  auto storeB = [&](int buf) {
    alignas(16) unsigned short tmp[16];
    *(uint4*)&tmp[0] = breg0;
    *(uint4*)&tmp[8] = breg1;
#pragma unroll
    for (int i = 0; i < 16; ++i) Bs[buf][(nb2 + i) * 40 + kr] = tmp[i];
  };

  const v8f vzero = {0, 0, 0, 0, 0, 0, 0, 0};
  v8f acc[4];
  acc[0] = acc[1] = acc[2] = acc[3] = vzero;

  issueA(0, 0);
  loadB(0);
  storeB(0);
  wait_async0();
  __syncthreads();

  for (int it = 0; it < 16; ++it) {
    const int cur = it & 1;
    if (it < 15) {
      issueA((it + 1) * 32, cur ^ 1);
      loadB((it + 1) * 32);
    }

    Frag a;
#pragma unroll
    for (int j = 0; j < 8; ++j)
      a.u[j] = *(const unsigned int*)&As[cur][(wave * 16 + lo) * 40 + kkA(j, hi)];
    Frag bf[4];
#pragma unroll
    for (int nb = 0; nb < 4; ++nb)
#pragma unroll
      for (int j = 0; j < 8; ++j)
        bf[nb].u[j] = *(const unsigned int*)&Bs[cur][(nb * 16 + lo) * 40 + kkB(j, hi)];
#pragma unroll
    for (int nb = 0; nb < 4; ++nb)
      acc[nb] = __builtin_amdgcn_wmma_f32_16x16x32_bf16(
          false, a.v, false, bf[nb].v, (short)0, acc[nb], false, false);

    if (it < 15) {
      storeB(cur ^ 1);
      wait_async0();
    }
    __syncthreads();
  }

#pragma unroll
  for (int nb = 0; nb < 4; ++nb) {
    int col = nbase + nb * 16 + lo;
    float bv = bias[col];
#pragma unroll
    for (int r = 0; r < 8; ++r) {
      int m = mbase + wave * 16 + hi * 8 + r;
      out[(size_t)m * DIM + col] = acc[nb][r] + bv;
    }
  }
}

// ---------------------------------------------------------------- launch
extern "C" void kernel_launch(void* const* d_in, const int* in_sizes, int n_in,
                              void* d_out, int out_size, void* d_ws, size_t ws_size,
                              hipStream_t stream) {
  const float* x      = (const float*)d_in[0];
  const float* qkv_w  = (const float*)d_in[1];
  const float* qkv_b  = (const float*)d_in[2];
  const float* proj_w = (const float*)d_in[3];
  const float* proj_b = (const float*)d_in[4];
  const float* gate_w = (const float*)d_in[5];
  const float* gate_b = (const float*)d_in[6];

  char* ws = (char*)d_ws;
  unsigned short* xb = (unsigned short*)(ws + 0);         //  8,388,608 B
  unsigned short* wq = (unsigned short*)(ws + 8388608);   //  1,572,864 B
  unsigned short* wp = (unsigned short*)(ws + 9961472);   //    524,288 B
  unsigned short* qb = (unsigned short*)(ws + 10485760);  //  8,388,608 B
  unsigned short* kb = (unsigned short*)(ws + 18874368);  //  8,388,608 B
  unsigned short* vb = (unsigned short*)(ws + 27262976);  //  8,388,608 B
  float*          gt = (float*)(ws + 35651584);           //    262,144 B
  unsigned short* yb = (unsigned short*)(ws + 35913728);  //  8,388,608 B

  convert_kernel<<<512, 256, 0, stream>>>(x, qkv_w, proj_w, xb, wq, wp);
  gate_kernel<<<256, 256, 0, stream>>>(x, gate_w, gate_b, gt);
  qkv_gemm_kernel<<<dim3(NC3 / 64, MROWS / 64), 128, 0, stream>>>(xb, wq, qkv_b, qb, kb, vb);
  attn_kernel<<<dim3(SEQ / 64, HEADS, BATCH), 128, 0, stream>>>(qb, kb, vb, gt, yb);
  proj_gemm_kernel<<<dim3(DIM / 64, MROWS / 64), 128, 0, stream>>>(yb, wp, proj_b, (float*)d_out);
}
